// BitwiseWavenet_35785667510800
// MI455X (gfx1250) — matrix operand order repaired and checked by reading, not compiled, not run on hardware
//
#include <hip/hip_runtime.h>
#include <cstdint>

// ---------------------------------------------------------------------------
// WaveNet fused for MI455X (gfx1250, wave32, WMMA, async LDS DMA).
//
// ~0.1 TFLOP vs 268 MB mandatory output write -> memory bound. Each block's 10
// dilated layers run fully in LDS (tile 512 + halo 512); all channel mixing is
// v_wmma_f32_16x16x32_f16. LDS buffers are position-major so every WMMA
// fragment is lane-contiguous (ds_load_b64/b128, not scalar u16 gathers).
// Inter-block residual is f16 position-major in HBM and moved with
// global_store_async_from_lds_b128 / global_load_async_to_lds_b128 + ASYNCcnt.
// ---------------------------------------------------------------------------

typedef __attribute__((ext_vector_type(16))) _Float16 v16h;
typedef __attribute__((ext_vector_type(4)))  _Float16 v4h;
typedef __attribute__((ext_vector_type(8)))  float    v8f;

#define WN_L       65536
#define WN_B       4
#define WN_RES     32
#define WN_LAYERS  10

#define TILE1  512
#define HALO1  512
#define BUF1   (TILE1 + 2 * HALO1)   // 1536
#define RP1    40                    // residS pitch (halves): 80B rows, 16B aligned
#define LP1    36                    // loutS pitch (halves): 72B rows, 8B aligned
#define SP1    36                    // skipS pitch (floats)
#define NT1    (BUF1 / 16)           // 96

#define PT2    256
#define NT2    (PT2 / 16)            // 16
#define HP2    264                   // head h pitch (halves): 528B rows

static __device__ __forceinline__ v8f wmma16(v16h a, v16h b, v8f c) {
  return __builtin_amdgcn_wmma_f32_16x16x32_f16(false, a, false, b, (short)0, c,
                                                false, false);
}

// A fragment 16x32 f16: row m, K contiguous [16*khi..+15].
static __device__ __forceinline__ v16h ldA(const _Float16* W, int pitch,
                                           int mbase, int lane) {
  const int khi = (lane >> 4) & 1;
  const _Float16* p = W + (mbase + (lane & 15)) * pitch + 16 * khi;
  const v4h c0 = *(const v4h*)(p);
  const v4h c1 = *(const v4h*)(p + 4);
  const v4h c2 = *(const v4h*)(p + 8);
  const v4h c3 = *(const v4h*)(p + 12);
  v16h a;
#pragma unroll
  for (int j = 0; j < 4; ++j) {
    a[j] = c0[j]; a[4 + j] = c1[j]; a[8 + j] = c2[j]; a[12 + j] = c3[j];
  }
  return a;
}

// B fragment 32x16 f16 from position-major LDS: 32 contiguous bytes per lane.
static __device__ __forceinline__ v16h ldB(const _Float16* S, int pitch,
                                           int col, int lane, int kofs = 0) {
  const _Float16* p = S + col * pitch + kofs + ((lane & 16) ? 16 : 0);
  const v4h c0 = *(const v4h*)(p);
  const v4h c1 = *(const v4h*)(p + 4);
  const v4h c2 = *(const v4h*)(p + 8);
  const v4h c3 = *(const v4h*)(p + 12);
  v16h b;
#pragma unroll
  for (int j = 0; j < 4; ++j) {
    b[j] = c0[j]; b[4 + j] = c1[j]; b[8 + j] = c2[j]; b[12 + j] = c3[j];
  }
  return b;
}

static __device__ __forceinline__ v8f bias_frag(const float* bias, int mbase,
                                                int lane) {
  const int khi = (lane >> 4) & 1;
  v8f c;
#pragma unroll
  for (int r = 0; r < 8; ++r) c[r] = bias[mbase + 8 * khi + r];
  return c;
}

static __device__ __forceinline__ uint32_t lds_off(const void* p) {
  return (uint32_t)(uintptr_t)p;   // flat LDS addr: low 32 bits = LDS offset
}

// ---------------------------------------------------------------------------
// Kernel 1: one WaveNet block fused in LDS. grid = (L/TILE1, B).
// resid16 (f16, [B][L][32]) carries the residual between the two blocks.
// ---------------------------------------------------------------------------
__global__ __launch_bounds__(256) void wavenet_block_kernel(
    int blk, const float* __restrict__ x, _Float16* __restrict__ resid16_,
    float* __restrict__ skip_acc,
    const float* __restrict__ w_start, const float* __restrict__ b_start,
    const float* __restrict__ filt_w, const float* __restrict__ filt_b,
    const float* __restrict__ gate_w, const float* __restrict__ gate_b,
    const float* __restrict__ res_w, const float* __restrict__ res_b,
    const float* __restrict__ skip_w, const float* __restrict__ skip_b) {
  extern __shared__ char smem[];
  _Float16* residS = (_Float16*)smem;                      // [BUF1][RP1]
  _Float16* loutS  = residS + BUF1 * RP1;                  // [BUF1][LP1]
  float*    skipS  = (float*)(loutS + BUF1 * LP1);         // [TILE1][SP1]
  _Float16* WfS    = (_Float16*)(skipS + TILE1 * SP1);     // [2][32][36]
  _Float16* WgS    = WfS + 2 * 32 * 36;
  _Float16* WrS    = WgS + 2 * 32 * 36;                    // [32][36]
  _Float16* WsS    = WrS + 32 * 36;
  float*    fbS    = (float*)(WsS + 32 * 36);
  float*    gbS    = fbS + 32;
  float*    rbS    = gbS + 32;
  float*    sbS    = rbS + 32;

  const int tid  = threadIdx.x;
  const int lane = tid & 31;
  const int wave = tid >> 5;
  const int bat  = blockIdx.y;
  const int tile = blockIdx.x;
  const long g0  = (long)tile * TILE1 - HALO1;
  const char* resid16 = (const char*)resid16_;

  // ---- seed residual tile (+halo) ----
  if (blk == 0) {
    for (int idx = tid; idx < BUF1 * 32; idx += 256) {
      const int p = idx >> 5, c = idx & 31;
      const long g = g0 + p;
      float v = 0.f;
      if (g >= 0 && g < WN_L)
        v = w_start[c] * x[(long)bat * WN_L + g] + b_start[c];
      residS[p * RP1 + c] = (_Float16)v;
    }
  } else {
    // async DMA: 4 x 16B chunks per position (rows are 16B aligned in LDS)
    for (int idx = tid; idx < BUF1 * 4; idx += 256) {
      const int p = idx >> 2, q = idx & 3;
      const long g = g0 + p;
      const uint32_t ldsa = lds_off(residS + p * RP1 + q * 8);
      if (g >= 0 && g < WN_L) {
        const uint64_t ga =
            (uint64_t)(resid16 + ((long)bat * WN_L + g) * 64 + q * 16);
        asm volatile("global_load_async_to_lds_b128 %0, %1, off"
                     :: "v"(ldsa), "v"(ga) : "memory");
      } else {
        *(int4*)(residS + p * RP1 + q * 8) = int4{0, 0, 0, 0};
      }
    }
    asm volatile("s_wait_asynccnt 0" ::: "memory");
  }
  for (int idx = tid; idx < TILE1 * 32; idx += 256)
    skipS[(idx >> 5) * SP1 + (idx & 31)] = 0.f;
  __syncthreads();

  for (int layer = 0; layer < WN_LAYERS; ++layer) {
    const int lw = blk * WN_LAYERS + layer;
    const float* fw = filt_w + (long)lw * 2048;   // [32][32][2]
    const float* gw = gate_w + (long)lw * 2048;
    const float* rw = res_w  + (long)lw * 1024;   // [32][32]
    const float* sw = skip_w + (long)lw * 1024;

    for (int idx = tid; idx < 2048; idx += 256) {   // -> [tap][o][36]
      const int t = idx >> 10, o = (idx & 1023) >> 5, i = idx & 31;
      WfS[t * (32 * 36) + o * 36 + i] = (_Float16)fw[(o * 32 + i) * 2 + t];
      WgS[t * (32 * 36) + o * 36 + i] = (_Float16)gw[(o * 32 + i) * 2 + t];
    }
    for (int idx = tid; idx < 1024; idx += 256) {
      const int o = idx >> 5, i = idx & 31;
      WrS[o * 36 + i] = (_Float16)rw[idx];
      WsS[o * 36 + i] = (_Float16)sw[idx];
    }
    if (tid < 32) {
      fbS[tid] = filt_b[(long)lw * 32 + tid];
      gbS[tid] = gate_b[(long)lw * 32 + tid];
      rbS[tid] = res_b [(long)lw * 32 + tid];
      sbS[tid] = skip_b[(long)lw * 32 + tid];
    }
    if (layer + 1 < WN_LAYERS) {   // global_prefetch_b8 path
      __builtin_prefetch(fw + 2048 + tid * 8, 0, 1);
      __builtin_prefetch(gw + 2048 + tid * 8, 0, 1);
    }
    __syncthreads();

    int off0, off1;
    if (layer == 0) { off0 = -1; off1 = 0; }
    else { off0 = -(1 << (layer - 1)); off1 = (1 << (layer - 1)); }

    // ---- Phase A: layer_out = (Wf * r + bf) .* (Wg * r + bg) ----
    for (int nt = wave; nt < NT1; nt += 8) {
      const int pos0 = nt * 16;
      const int n    = lane & 15;
      const int khi  = (lane >> 4) & 1;
      int c0 = pos0 + n + off0; c0 = c0 < 0 ? 0 : (c0 > BUF1 - 1 ? BUF1 - 1 : c0);
      int c1 = pos0 + n + off1; c1 = c1 < 0 ? 0 : (c1 > BUF1 - 1 ? BUF1 - 1 : c1);
      const v16h b0 = ldB(residS, RP1, c0, lane);
      const v16h b1 = ldB(residS, RP1, c1, lane);
#pragma unroll
      for (int mt = 0; mt < 2; ++mt) {
        const int mbase = mt * 16;
        v8f f = bias_frag(fbS, mbase, lane);
        f = wmma16(ldA(WfS,           36, mbase, lane), b0, f);
        f = wmma16(ldA(WfS + 32 * 36, 36, mbase, lane), b1, f);
        v8f g = bias_frag(gbS, mbase, lane);
        g = wmma16(ldA(WgS,           36, mbase, lane), b0, g);
        g = wmma16(ldA(WgS + 32 * 36, 36, mbase, lane), b1, g);
        _Float16* dp = loutS + (pos0 + n) * LP1 + mbase + 8 * khi;
        v4h lo, hi;
#pragma unroll
        for (int r = 0; r < 4; ++r) {
          lo[r] = (_Float16)(f[r] * g[r]);
          hi[r] = (_Float16)(f[r + 4] * g[r + 4]);
        }
        *(v4h*)(dp) = lo; *(v4h*)(dp + 4) = hi;
      }
    }
    __syncthreads();

    // ---- Phase B: r += Wr*lo + br ; skip += Ws*lo + bs (central tile) ----
    for (int nt = wave; nt < NT1; nt += 8) {
      const int pos0 = nt * 16;
      const int n    = lane & 15;
      const int khi  = (lane >> 4) & 1;
      const v16h bL  = ldB(loutS, LP1, pos0 + n, lane);
      const int pp   = pos0 + n;
      const int ct   = pp - HALO1;
#pragma unroll
      for (int mt = 0; mt < 2; ++mt) {
        const int mbase = mt * 16;
        v8f rr = bias_frag(rbS, mbase, lane);
        rr = wmma16(ldA(WrS, 36, mbase, lane), bL, rr);
        v8f ss = bias_frag(sbS, mbase, lane);
        ss = wmma16(ldA(WsS, 36, mbase, lane), bL, ss);
        _Float16* rp = residS + pp * RP1 + mbase + 8 * khi;
        v4h o0 = *(v4h*)(rp), o1 = *(v4h*)(rp + 4);
#pragma unroll
        for (int r = 0; r < 4; ++r) {
          o0[r] = (_Float16)((float)o0[r] + rr[r]);
          o1[r] = (_Float16)((float)o1[r] + rr[r + 4]);
        }
        *(v4h*)(rp) = o0; *(v4h*)(rp + 4) = o1;
        if (ct >= 0 && ct < TILE1) {
          float* sp = skipS + ct * SP1 + mbase + 8 * khi;
#pragma unroll
          for (int r = 0; r < 8; ++r) sp[r] += ss[r];
        }
      }
    }
    __syncthreads();
  }

  // ---- drain: skip (position-major f32) + residual (async f16 DMA, blk 0) ----
  for (int idx = tid; idx < TILE1 * 32; idx += 256) {
    const int t = idx >> 5, c = idx & 31;
    const long gi = ((long)bat * WN_L + (long)tile * TILE1 + t) * 32 + c;
    const float sv = skipS[t * SP1 + c];
    if (blk == 0) skip_acc[gi] = sv; else skip_acc[gi] += sv;
  }
  if (blk == 0) {
    char* r16 = (char*)resid16_;
    for (int idx = tid; idx < TILE1 * 4; idx += 256) {
      const int t = idx >> 2, q = idx & 3;
      const uint32_t ldsa = lds_off(residS + (HALO1 + t) * RP1 + q * 8);
      const uint64_t ga =
          (uint64_t)(r16 + ((long)bat * WN_L + (long)tile * TILE1 + t) * 64 +
                     q * 16);
      asm volatile("global_store_async_from_lds_b128 %0, %1, off"
                   :: "v"(ga), "v"(ldsa) : "memory");
    }
    asm volatile("s_wait_asynccnt 0" ::: "memory");
  }
}

// ---------------------------------------------------------------------------
// Kernel 2: head. out = W2 * relu(W1 * relu(skip) + b1) + b2. grid=(L/PT2, B).
// skip_g is position-major -> each lane's skip fragment is 4 x b128 loads.
// ---------------------------------------------------------------------------
__global__ __launch_bounds__(256) void wavenet_head_kernel(
    const float* __restrict__ skip_g, const float* __restrict__ w1,
    const float* __restrict__ bb1, const float* __restrict__ w2,
    const float* __restrict__ bb2, float* __restrict__ out) {
  extern __shared__ char smem[];
  _Float16* W1S = (_Float16*)smem;            // [256][36]
  _Float16* W2S = W1S + 256 * 36;             // [256][264]
  _Float16* hS  = W2S + 256 * 264;            // 8 waves * [16][264]
  float*    b1S = (float*)(hS + 8 * 16 * HP2);
  float*    b2S = b1S + 256;

  const int tid  = threadIdx.x;
  const int lane = tid & 31;
  const int wave = tid >> 5;
  const int bat  = blockIdx.y;
  const long l0b = (long)blockIdx.x * PT2;

  for (int idx = tid; idx < 256 * 32; idx += 256)
    W1S[(idx >> 5) * 36 + (idx & 31)] = (_Float16)w1[idx];
  for (int idx = tid; idx < 256 * 256; idx += 256)
    W2S[(idx >> 8) * HP2 + (idx & 255)] = (_Float16)w2[idx];
  if (tid < 256) { b1S[tid] = bb1[tid]; b2S[tid] = bb2[tid]; }
  __syncthreads();

  _Float16* hW  = hS + wave * (16 * HP2);     // [n][HP2]
  const int n   = lane & 15;
  const int khi = (lane >> 4) & 1;

  for (int nt = wave; nt < NT2; nt += 8) {
    const long lp = l0b + nt * 16 + n;

    // relu(skip) B fragment: 16 contiguous floats per lane
    const float* sp = skip_g + ((long)bat * WN_L + lp) * 32 + khi * 16;
    v16h sB;
    {
      const float4 f0 = *(const float4*)(sp);
      const float4 f1 = *(const float4*)(sp + 4);
      const float4 f2 = *(const float4*)(sp + 8);
      const float4 f3 = *(const float4*)(sp + 12);
      sB[0] = (_Float16)fmaxf(f0.x, 0.f); sB[1] = (_Float16)fmaxf(f0.y, 0.f);
      sB[2] = (_Float16)fmaxf(f0.z, 0.f); sB[3] = (_Float16)fmaxf(f0.w, 0.f);
      sB[4] = (_Float16)fmaxf(f1.x, 0.f); sB[5] = (_Float16)fmaxf(f1.y, 0.f);
      sB[6] = (_Float16)fmaxf(f1.z, 0.f); sB[7] = (_Float16)fmaxf(f1.w, 0.f);
      sB[8] = (_Float16)fmaxf(f2.x, 0.f); sB[9] = (_Float16)fmaxf(f2.y, 0.f);
      sB[10] = (_Float16)fmaxf(f2.z, 0.f); sB[11] = (_Float16)fmaxf(f2.w, 0.f);
      sB[12] = (_Float16)fmaxf(f3.x, 0.f); sB[13] = (_Float16)fmaxf(f3.y, 0.f);
      sB[14] = (_Float16)fmaxf(f3.z, 0.f); sB[15] = (_Float16)fmaxf(f3.w, 0.f);
    }
    __builtin_prefetch(sp + 128 * 32, 0, 1);   // this wave's next tile

    // h = relu(W1 * s + b1) -> wave-private LDS slab
    for (int mt = 0; mt < 16; ++mt) {
      v8f acc = bias_frag(b1S, mt * 16, lane);
      acc = wmma16(ldA(W1S, 36, mt * 16, lane), sB, acc);
      _Float16* hp = hW + n * HP2 + mt * 16 + 8 * khi;
      v4h lo, hi;
#pragma unroll
      for (int r = 0; r < 4; ++r) {
        lo[r] = (_Float16)fmaxf(acc[r], 0.f);
        hi[r] = (_Float16)fmaxf(acc[r + 4], 0.f);
      }
      *(v4h*)(hp) = lo; *(v4h*)(hp + 4) = hi;
    }

    // out = W2 * h + b2   (K = 256 -> 8 chained WMMAs per output tile)
    for (int mt = 0; mt < 16; ++mt) {
      v8f acc = bias_frag(b2S, mt * 16, lane);
#pragma unroll
      for (int ks = 0; ks < 8; ++ks) {
        const v16h aW = ldA(W2S + ks * 32, HP2, mt * 16, lane);
        const v16h hB = ldB(hW, HP2, n, lane, ks * 32);
        acc = wmma16(aW, hB, acc);
      }
#pragma unroll
      for (int r = 0; r < 8; ++r) {
        const int m = mt * 16 + 8 * khi + r;
        out[((long)bat * 256 + m) * WN_L + lp] = acc[r];
      }
    }
  }
}

// ---------------------------------------------------------------------------
extern "C" void kernel_launch(void* const* d_in, const int* in_sizes, int n_in,
                              void* d_out, int out_size, void* d_ws,
                              size_t ws_size, hipStream_t stream) {
  (void)in_sizes; (void)n_in; (void)out_size; (void)ws_size;
  const float* x       = (const float*)d_in[0];
  const float* w_start = (const float*)d_in[1];
  const float* b_start = (const float*)d_in[2];
  const float* filt_w  = (const float*)d_in[3];
  const float* filt_b  = (const float*)d_in[4];
  const float* gate_w  = (const float*)d_in[5];
  const float* gate_b  = (const float*)d_in[6];
  const float* res_w   = (const float*)d_in[7];
  const float* res_b   = (const float*)d_in[8];
  const float* skip_w  = (const float*)d_in[9];
  const float* skip_b  = (const float*)d_in[10];
  const float* w_end1  = (const float*)d_in[11];
  const float* b_end1  = (const float*)d_in[12];
  const float* w_end2  = (const float*)d_in[13];
  const float* b_end2  = (const float*)d_in[14];
  float* out = (float*)d_out;

  // ws: resid16 (f16 [B][L][32], 16.8 MB) then skip_g (f32 [B][L][32], 33.6 MB)
  _Float16* resid16 = (_Float16*)d_ws;
  float*    skip_g  = (float*)((char*)d_ws +
                               (size_t)WN_B * WN_L * 32 * sizeof(_Float16));

  constexpr size_t SMEM1 = (size_t)BUF1 * RP1 * 2 + (size_t)BUF1 * LP1 * 2 +
                           (size_t)TILE1 * SP1 * 4 +
                           (size_t)(2 * 2 * 32 * 36 + 2 * 32 * 36) * 2 +
                           (size_t)4 * 32 * 4;                     // 321,536 B
  constexpr size_t SMEM2 = (size_t)(256 * 36 + 256 * HP2 + 8 * 16 * HP2) * 2 +
                           (size_t)2 * 256 * 4;                    // ~223 KB

  dim3 g1(WN_L / TILE1, WN_B);  // 128 x 4
  wavenet_block_kernel<<<g1, 256, SMEM1, stream>>>(
      0, x, resid16, skip_g, w_start, b_start, filt_w, filt_b, gate_w, gate_b,
      res_w, res_b, skip_w, skip_b);
  wavenet_block_kernel<<<g1, 256, SMEM1, stream>>>(
      1, x, resid16, skip_g, w_start, b_start, filt_w, filt_b, gate_w, gate_b,
      res_w, res_b, skip_w, skip_b);

  dim3 g2(WN_L / PT2, WN_B);    // 256 x 4
  wavenet_head_kernel<<<g2, 256, SMEM2, stream>>>(skip_g, w_end1, b_end1,
                                                  w_end2, b_end2, out);
}
